// Decoder_16054587752897
// MI455X (gfx1250) — matrix operand order repaired and checked by reading, not compiled, not run on hardware
//
#include <hip/hip_runtime.h>
#include <hip/hip_bf16.h>
#include <math.h>

// MI455X / gfx1250: wave32, WMMA 16x16x32 f16 path.
typedef __attribute__((ext_vector_type(16))) _Float16 v16h;
typedef __attribute__((ext_vector_type(8)))  float    v8f;

#define N_B   8
#define C_IN  96
#define L_SEQ 512
#define D_ATT 32
#define FF_H  384

// ---------------- WMMA operand loaders (per CDNA5 ISA VGPR layouts) ----------------
// 16-bit A-matrix 16x32 (MxK): lanes 0-15 -> M=lane, K-halves {0..7,16..23};
// lanes 16-31 -> M=lane-16, K-halves {8..15,24..31}.
__device__ __forceinline__ void load_a16(const _Float16* __restrict__ a_lds, int ldk,
                                         int k0, int lane, v16h& a) {
  int m   = lane & 15;
  int khi = (lane & 16) ? 8 : 0;
  #pragma unroll
  for (int e = 0; e < 16; ++e) {
    int v = e >> 1, odd = e & 1;
    int kk = (v < 4) ? (khi + 2*v + odd) : (16 + khi + 2*(v - 4) + odd);
    a[e] = a_lds[m * ldk + k0 + kk];
  }
}

// 16-bit B-matrix 32x16 (KxN): lanes 0-15 -> N=lane, K-halves {0..7,16..23};
// lanes 16-31 -> N=lane-16, K-halves {8..15,24..31} (same K order as A).
// Weight stored row-major W[n][k] (torch Linear): B[k][n] = W[n*Kdim + k].
__device__ __forceinline__ void load_b16_Wnk(const float* __restrict__ W, int Kdim,
                                             int n0, int k0, int lane, v16h& b) {
  int nn = n0 + (lane & 15);
  int khi = (lane & 16) ? 8 : 0;
  #pragma unroll
  for (int e = 0; e < 16; ++e) {
    int kk = k0 + ((e < 8) ? (khi + e) : (16 + khi + (e - 8)));
    b[e] = (_Float16)W[nn * Kdim + kk];
  }
}
// Weight stored (K, N) row-major (attention Wt/Wx are (96,32), q = xt @ Wt).
__device__ __forceinline__ void load_b16_Wkn(const float* __restrict__ W, int Ndim,
                                             int n0, int k0, int lane, v16h& b) {
  int nn = n0 + (lane & 15);
  int khi = (lane & 16) ? 8 : 0;
  #pragma unroll
  for (int e = 0; e < 16; ++e) {
    int kk = k0 + ((e < 8) ? (khi + e) : (16 + khi + (e - 8)));
    b[e] = (_Float16)W[kk * Ndim + nn];
  }
}

// ---------------- Kernel 1: q/k projections via WMMA ----------------
// grid = 256 blocks of 16 rows; 64 threads (2 waves): wave0 -> q (Wt), wave1 -> k (Wx).
__global__ void qk_proj_kernel(const float* __restrict__ x,   // (N, C, L)
                               const float* __restrict__ Wx,  // (96, 32)
                               const float* __restrict__ Wt,  // (96, 32)
                               float* __restrict__ q,         // (N, L, 32)
                               float* __restrict__ kb) {      // (N, L, 32)
  __shared__ _Float16 a16[16 * C_IN];
  int r0 = blockIdx.x * 16;
  int n = r0 / L_SEQ, l0 = r0 % L_SEQ;
  int t = threadIdx.x;
  for (int idx = t; idx < 16 * C_IN; idx += 64) {
    int m = idx / C_IN, c = idx % C_IN;                      // xt[m][c] = x[n][c][l0+m]
    a16[idx] = (_Float16)x[(n * C_IN + c) * L_SEQ + (l0 + m)];
  }
  __syncthreads();
  int wave = t >> 5, lane = t & 31;
  const float* W = wave ? Wx : Wt;
  float* outp    = wave ? kb : q;
  #pragma unroll
  for (int nt = 0; nt < 2; ++nt) {
    v8f acc = {};
    #pragma unroll
    for (int kt = 0; kt < 3; ++kt) {                          // K = 96 = 3*32
      v16h a, b;
      load_a16(a16, C_IN, kt * 32, lane, a);
      load_b16_Wkn(W, D_ATT, nt * 16, kt * 32, lane, b);
      acc = __builtin_amdgcn_wmma_f32_16x16x32_f16(false, a, false, b, (short)0, acc, false, false);
    }
    int Nc = nt * 16 + (lane & 15);
    int Mb = (lane & 16) ? 8 : 0;                             // D: vgpr r -> M=r (+8 for hi lanes)
    #pragma unroll
    for (int r = 0; r < 8; ++r) outp[(r0 + r + Mb) * D_ATT + Nc] = acc[r];
  }
}

// ---------------- Kernel 2: additive attention + residual + LN0 ----------------
// One 64-thread block per (n, i). Full-row e for the faithful full-row max,
// band (j-i in [-32,31]) softmax with +1e-6 denom, v = a @ xt, then LayerNorm.
__global__ void attn_ln_kernel(const float* __restrict__ x, const float* __restrict__ q,
                               const float* __restrict__ kbuf, const float* __restrict__ bh,
                               const float* __restrict__ Wa, const float* __restrict__ ba,
                               const float* __restrict__ g0, const float* __restrict__ b0,
                               float* __restrict__ x2) {      // (N, L, 96)
  __shared__ float qb[32], wa[32], ev[L_SEQ], red[64], pband[64], rvec[C_IN], stats[2];
  int bi = blockIdx.x;
  int n = bi / L_SEQ, i = bi % L_SEQ;
  int t = threadIdx.x;
  if (t < 32) { qb[t] = q[(n * L_SEQ + i) * D_ATT + t] + bh[t]; wa[t] = Wa[t]; }
  __syncthreads();
  float lmax = -1e30f;
  #pragma unroll
  for (int jj = 0; jj < 8; ++jj) {
    int j = t + jj * 64;
    const float* kr = kbuf + (n * L_SEQ + j) * D_ATT;
    float e = ba[0];
    #pragma unroll
    for (int d = 0; d < 32; ++d) e += wa[d] * tanhf(qb[d] + kr[d]);
    ev[j] = e;
    lmax = fmaxf(lmax, e);
  }
  red[t] = lmax;
  __syncthreads();
  for (int s = 32; s > 0; s >>= 1) { if (t < s) red[t] = fmaxf(red[t], red[t + s]); __syncthreads(); }
  float emax = red[0];
  __syncthreads();
  int jlo = max(0, i - 32), jhi = min(L_SEQ - 1, i + 31);
  int len = jhi - jlo + 1;
  float p = (t < len) ? expf(ev[jlo + t] - emax) : 0.f;
  red[t] = p;
  __syncthreads();
  for (int s = 32; s > 0; s >>= 1) { if (t < s) red[t] += red[t + s]; __syncthreads(); }
  float denom = red[0] + 1e-6f;
  pband[t] = p / denom;
  __syncthreads();
  for (int c = t; c < C_IN; c += 64) {
    const float* xr = x + (n * C_IN + c) * L_SEQ;
    float v = 0.f;
    for (int tt = 0; tt < len; ++tt) v += pband[tt] * xr[jlo + tt];
    rvec[c] = v + xr[i];                                      // residual
  }
  __syncthreads();
  if (t == 0) {
    float s = 0.f, s2 = 0.f;
    for (int c = 0; c < C_IN; ++c) { s += rvec[c]; s2 += rvec[c] * rvec[c]; }
    float mu = s / (float)C_IN;
    stats[0] = mu;
    stats[1] = rsqrtf(fmaxf(s2 / (float)C_IN - mu * mu, 0.f) + 1e-5f);
  }
  __syncthreads();
  float mu = stats[0], rs = stats[1];
  for (int c = t; c < C_IN; c += 64)
    x2[(n * L_SEQ + i) * C_IN + c] = (rvec[c] - mu) * rs * g0[c] + b0[c];
}

// ---------------- Kernel 3: fused FFN (WMMA) + residual + LN1 ----------------
// 256 blocks x 16 rows; 256 threads (8 waves). GEMM1: 24 n-tiles (3/wave), K=96.
// GEMM2: 6 n-tiles (waves 0-5), K=384. Then per-row LayerNorm.
__global__ void ffn_ln_kernel(const float* __restrict__ x2,
                              const float* __restrict__ W0, const float* __restrict__ b0,
                              const float* __restrict__ W1, const float* __restrict__ b1,
                              const float* __restrict__ g1, const float* __restrict__ bb1,
                              float* __restrict__ x4) {       // (N, L, 96)
  __shared__ _Float16 a16[16 * C_IN];
  __shared__ float    a32[16 * C_IN];
  __shared__ _Float16 hid[16 * FF_H];
  __shared__ float    r3[16 * C_IN];
  int r0 = blockIdx.x * 16;
  int t = threadIdx.x;
  __builtin_prefetch(&W0[t * 144], 0, 3);                     // -> global_prefetch_b8
  __builtin_prefetch(&W1[t * 144], 0, 3);
  for (int idx = t; idx < 16 * C_IN; idx += 256) {
    float v = x2[r0 * C_IN + idx];
    a32[idx] = v; a16[idx] = (_Float16)v;
  }
  __syncthreads();
  int wave = t >> 5, lane = t & 31;
  #pragma unroll
  for (int w = 0; w < 3; ++w) {
    int nt = wave * 3 + w;
    v8f acc = {};
    #pragma unroll
    for (int kt = 0; kt < 3; ++kt) {
      v16h a, b;
      load_a16(a16, C_IN, kt * 32, lane, a);
      load_b16_Wnk(W0, C_IN, nt * 16, kt * 32, lane, b);      // hid[m,o] = sum_k A[m,k]*W0[o,k]
      acc = __builtin_amdgcn_wmma_f32_16x16x32_f16(false, a, false, b, (short)0, acc, false, false);
    }
    int Nc = nt * 16 + (lane & 15);
    int Mb = (lane & 16) ? 8 : 0;
    float bias = b0[Nc];
    #pragma unroll
    for (int r = 0; r < 8; ++r)
      hid[(r + Mb) * FF_H + Nc] = (_Float16)fmaxf(acc[r] + bias, 0.f);
  }
  __syncthreads();
  if (wave < 6) {
    int nt = wave;
    v8f acc = {};
    for (int kt = 0; kt < 12; ++kt) {
      v16h a, b;
      load_a16(hid, FF_H, kt * 32, lane, a);
      load_b16_Wnk(W1, FF_H, nt * 16, kt * 32, lane, b);
      acc = __builtin_amdgcn_wmma_f32_16x16x32_f16(false, a, false, b, (short)0, acc, false, false);
    }
    int Nc = nt * 16 + (lane & 15);
    int Mb = (lane & 16) ? 8 : 0;
    float bias = b1[Nc];
    #pragma unroll
    for (int r = 0; r < 8; ++r) {
      int m = r + Mb;
      r3[m * C_IN + Nc] = acc[r] + bias + a32[m * C_IN + Nc]; // residual
    }
  }
  __syncthreads();
  if (t < 16) {
    const float* rr = &r3[t * C_IN];
    float s = 0.f, s2 = 0.f;
    for (int c = 0; c < C_IN; ++c) { s += rr[c]; s2 += rr[c] * rr[c]; }
    float mu = s / (float)C_IN;
    float rs = rsqrtf(fmaxf(s2 / (float)C_IN - mu * mu, 0.f) + 1e-5f);
    float* outp = x4 + (r0 + t) * C_IN;
    for (int c = 0; c < C_IN; ++c) outp[c] = (rr[c] - mu) * rs * g1[c] + bb1[c];
  }
}

// ---------------- Kernels 4-6: upsample x2 (nearest, crop) + conv1d K=7 pad(3,3) + ReLU ----
__global__ void up_conv_relu_kernel(const float* __restrict__ src, const float* __restrict__ w,
                                    const float* __restrict__ bias, float* __restrict__ dst,
                                    int Cin, int Cout, int Lout, int srcNLC, int Lin) {
  int idx = blockIdx.x * blockDim.x + threadIdx.x;
  int total = N_B * Cout * Lout;
  if (idx >= total) return;
  int l  = idx % Lout;
  int co = (idx / Lout) % Cout;
  int n  = idx / (Lout * Cout);
  const float* wr = w + co * Cin * 7;
  float acc = bias[co];
  for (int ci = 0; ci < Cin; ++ci) {
    #pragma unroll
    for (int kk = 0; kk < 7; ++kk) {
      int p = l - 3 + kk;                                     // position in upsampled seq
      if (p < 0 || p >= Lout) continue;
      int ps = p >> 1;                                        // nearest-neighbor source
      float s = srcNLC ? src[(n * Lin + ps) * Cin + ci]       // x4 is (N, L, C)
                       : src[(n * Cin + ci) * Lin + ps];
      acc += wr[ci * 7 + kk] * s;
    }
  }
  dst[idx] = fmaxf(acc, 0.f);
}

// ---------------- Kernel 7: final conv1d K=11 pad(5,5) + sigmoid ----------------
__global__ void out_conv_sigmoid_kernel(const float* __restrict__ src, const float* __restrict__ w,
                                        const float* __restrict__ bias, float* __restrict__ dst) {
  int idx = blockIdx.x * blockDim.x + threadIdx.x;
  if (idx >= N_B * 4096) return;
  int l = idx % 4096, n = idx / 4096;
  float acc = bias[0];
  for (int ci = 0; ci < 32; ++ci) {
    const float* sr = src + (n * 32 + ci) * 4096;
    #pragma unroll
    for (int kk = 0; kk < 11; ++kk) {
      int p = l - 5 + kk;
      if (p >= 0 && p < 4096) acc += w[ci * 11 + kk] * sr[p];
    }
  }
  dst[idx] = 1.f / (1.f + expf(-acc));
}

extern "C" void kernel_launch(void* const* d_in, const int* in_sizes, int n_in,
                              void* d_out, int out_size, void* d_ws, size_t ws_size,
                              hipStream_t stream) {
  (void)in_sizes; (void)n_in; (void)out_size; (void)ws_size;
  const float* x     = (const float*)d_in[0];
  const float* Wx    = (const float*)d_in[1];
  const float* Wt    = (const float*)d_in[2];
  const float* bh    = (const float*)d_in[3];
  const float* Wa    = (const float*)d_in[4];
  const float* ba    = (const float*)d_in[5];
  const float* ln0_g = (const float*)d_in[6];
  const float* ln0_b = (const float*)d_in[7];
  const float* ff_w0 = (const float*)d_in[8];
  const float* ff_b0 = (const float*)d_in[9];
  const float* ff_w1 = (const float*)d_in[10];
  const float* ff_b1 = (const float*)d_in[11];
  const float* ln1_g = (const float*)d_in[12];
  const float* ln1_b = (const float*)d_in[13];
  const float* up_w0 = (const float*)d_in[14];
  const float* up_b0 = (const float*)d_in[15];
  const float* up_w1 = (const float*)d_in[16];
  const float* up_b1 = (const float*)d_in[17];
  const float* up_w2 = (const float*)d_in[18];
  const float* up_b2 = (const float*)d_in[19];
  const float* out_w = (const float*)d_in[20];
  const float* out_b = (const float*)d_in[21];

  // workspace layout (floats); total ~13.6 MB
  float* ws = (float*)d_ws;
  float* q  = ws;                         // 8*512*32
  float* kb = q  + N_B * L_SEQ * D_ATT;   // 8*512*32
  float* x2 = kb + N_B * L_SEQ * D_ATT;   // 8*512*96
  float* x4 = x2 + N_B * L_SEQ * C_IN;    // 8*512*96
  float* h1 = x4 + N_B * L_SEQ * C_IN;    // 8*64*1024
  float* h2 = h1 + N_B * 64 * 1024;       // 8*48*2048
  float* h3 = h2 + N_B * 48 * 2048;       // 8*32*4096

  qk_proj_kernel<<<(N_B * L_SEQ) / 16, 64, 0, stream>>>(x, Wx, Wt, q, kb);
  attn_ln_kernel<<<N_B * L_SEQ, 64, 0, stream>>>(x, q, kb, bh, Wa, ba, ln0_g, ln0_b, x2);
  ffn_ln_kernel<<<(N_B * L_SEQ) / 16, 256, 0, stream>>>(x2, ff_w0, ff_b0, ff_w1, ff_b1, ln1_g, ln1_b, x4);
  up_conv_relu_kernel<<<(N_B * 64 * 1024 + 255) / 256, 256, 0, stream>>>(x4, up_w0, up_b0, h1, 96, 64, 1024, 1, 512);
  up_conv_relu_kernel<<<(N_B * 48 * 2048 + 255) / 256, 256, 0, stream>>>(h1, up_w1, up_b1, h2, 64, 48, 2048, 0, 1024);
  up_conv_relu_kernel<<<(N_B * 32 * 4096 + 255) / 256, 256, 0, stream>>>(h2, up_w2, up_b2, h3, 48, 32, 4096, 0, 2048);
  out_conv_sigmoid_kernel<<<(N_B * 4096 + 255) / 256, 256, 0, stream>>>(h3, out_w, out_b, (float*)d_out);
}
